// PointTransformerDecoderInterp_39565238730825
// MI455X (gfx1250) — matrix operand order repaired and checked by reading, not compiled, not run on hardware
//
#include <hip/hip_runtime.h>
#include <hip/hip_bf16.h>
#include <cstdint>

typedef __attribute__((ext_vector_type(16))) _Float16 v16h;
typedef __attribute__((ext_vector_type(8)))  float    v8f;
typedef __attribute__((ext_vector_type(4)))  int      v4i;

#define BB    2
#define NQ    32768
#define NA    1024
#define DIM   256
#define HH    50
#define HP    64
#define NBLK  5
#define VAR_INV 25.0f   // 1 / (0.2^2)

// CDNA5 async memory->LDS path (ASYNCcnt, CDNA5 ISA).
// Guarded: falls back to register-prefetch staging if the builtins are absent.
#if defined(__HIP_DEVICE_COMPILE__) && defined(__has_builtin)
# if __has_builtin(__builtin_amdgcn_global_load_async_to_lds_b128) && \
     __has_builtin(__builtin_amdgcn_s_wait_asynccnt)
#  define USE_ASYNC_LDS 1
# endif
#endif
#ifndef USE_ASYNC_LDS
# define USE_ASYNC_LDS 0
#endif

#if USE_ASYNC_LDS
typedef __attribute__((address_space(1))) v4i gv4i;   // global int4
typedef __attribute__((address_space(3))) v4i lv4i;   // LDS int4
// cooperative async copy of one 16KB tile (1024 uint4) by 256 threads
__device__ __forceinline__ void async_stage16k(const uint4* g, uint4* l, int tid) {
#pragma unroll
  for (int i = 0; i < 4; ++i)
    __builtin_amdgcn_global_load_async_to_lds_b128(
        (gv4i*)(g + tid * 4 + i), (lv4i*)(l + tid * 4 + i), 0, 0);
}
#endif

// ---------------------------------------------------------------------------
// Fragment loaders (layouts per CDNA5 ISA 7.12.2).
// A-frag (16-bit 16x32): lane L -> row M=L%16; pair start
//   = (L>=16 ? 8:0) + (j<4 ? 2j : 8+2j)  for dword j.
// ---------------------------------------------------------------------------
__device__ __forceinline__ v16h load_afrag(const _Float16* __restrict__ base,
                                           int stride_h, int k0, int lane) {
  int m  = lane & 15;
  int kb = (lane & 16) ? 8 : 0;
  const uint32_t* p = reinterpret_cast<const uint32_t*>(base + (size_t)m * stride_h + k0);
  union { v16h h; uint32_t u[8]; } t;
#pragma unroll
  for (int j = 0; j < 8; ++j) {
    int kp = kb + ((j < 4) ? (2 * j) : (8 + 2 * j));
    t.u[j] = p[kp >> 1];
  }
  return t.h;
}

// B-frag (16-bit 32x16): lane = column N; lanes 0-15 hold K=0..7,16..23, lanes
// 16-31 hold K=8..15,24..31.  Tiles are pre-swizzled [kt][n][32] so each lane
// reads two 8-half chunks (two b128 loads) -- works for global or LDS pointers.
__device__ __forceinline__ v16h load_bfrag(const _Float16* __restrict__ Wsw,
                                           int nt, int Np, int lane) {
  int n = nt * 16 + (lane & 15);
  const uint4* p = reinterpret_cast<const uint4*>(
      Wsw + (((size_t)n) << 5) + ((lane & 16) ? 8 : 0));
  union { v16h h; uint4 u[2]; } t;
  t.u[0] = p[0];
  t.u[1] = p[2];
  return t.h;
}

__device__ __forceinline__ v16h relu_frag(v16h a) {
  v16h r;
#pragma unroll
  for (int i = 0; i < 16; ++i) {
    _Float16 x = a[i];
    r[i] = (x > (_Float16)0) ? x : (_Float16)0;
  }
  return r;
}

// ---------------------------------------------------------------------------
// Staged GEMM: all waves in the block share one LDS tile of B.  Next tile is
// prefetched into registers while the current one is consumed; single LDS
// buffer with two barriers per tile (uniform control flow => legal).
// ---------------------------------------------------------------------------
template <int KT, int NT, int NP, int NTHR, bool RELU>
__device__ __forceinline__ void gemm_staged(const _Float16* __restrict__ W,
                                            const v16h* a, v8f* acc,
                                            _Float16* sbuf, int tid, int lane) {
  constexpr int TILE_U4 = (32 * NP) / 8;     // uint4 per tile
  constexpr int PT = TILE_U4 / NTHR;         // uint4 per thread
  const uint4* gw = reinterpret_cast<const uint4*>(W);
  uint4* sb4 = reinterpret_cast<uint4*>(sbuf);
  uint4 pf[PT];
#pragma unroll
  for (int i = 0; i < PT; ++i) pf[i] = gw[tid * PT + i];
#pragma unroll
  for (int kt = 0; kt < KT; ++kt) {
    __syncthreads();                          // previous tile fully consumed
#pragma unroll
    for (int i = 0; i < PT; ++i) sb4[tid * PT + i] = pf[i];
    if (kt + 1 < KT) {
#pragma unroll
      for (int i = 0; i < PT; ++i)
        pf[i] = gw[(size_t)(kt + 1) * TILE_U4 + tid * PT + i];
    }
    __syncthreads();                          // stores visible to all waves
    v16h af = RELU ? relu_frag(a[kt]) : a[kt];
#pragma unroll
    for (int nt = 0; nt < NT; ++nt) {
      v16h bf = load_bfrag(sbuf, nt, NP, lane);
      acc[nt] = __builtin_amdgcn_wmma_f32_16x16x32_f16(
          false, af, false, bf, (short)0, acc[nt], false, false);
    }
  }
}

// ---------------------------------------------------------------------------
// Prep: pad + swizzle a row-major [Ks x Ns] f32 matrix into f16 [kt][n][32].
// ---------------------------------------------------------------------------
__global__ void swz_pad_kernel(const float* __restrict__ src, _Float16* __restrict__ dst,
                               int Ks, int Ns, int Kp, int Np) {
  int idx = blockIdx.x * blockDim.x + threadIdx.x;
  if (idx >= Kp * Np) return;
  int k = idx / Np, n = idx - k * Np;
  float v = (k < Ks && n < Ns) ? src[(size_t)k * Ns + n] : 0.0f;
  dst[((((size_t)(k >> 5)) * Np + n) << 5) + (k & 31)] = (_Float16)v;
}

__global__ void pad_bias_kernel(const float* __restrict__ src, float* __restrict__ dst,
                                int Ns, int Np) {
  int n = blockIdx.x * blockDim.x + threadIdx.x;
  if (n < Np) dst[n] = (n < Ns) ? src[n] : 0.0f;
}

// ---------------------------------------------------------------------------
// Phase 1: gaussian-softmax interpolation fused with weight @ feats GEMM.
// 8 waves/block share staged anchor-feature tiles (async double-buffered when
// the gfx1250 async-LDS builtins are available); 1 wave = 16 queries.
// exp(-d^2/VAR) via ||q||^2+||a||^2-2q.a; logits<=0 so no max-shift needed;
// accumulators normalized by 1/sum afterwards.
// ---------------------------------------------------------------------------
__global__ __launch_bounds__(256) void interp_kernel(
    const float* __restrict__ xyz_q, const float* __restrict__ anchors,
    const _Float16* __restrict__ af_swz, _Float16* __restrict__ c_out) {
  __shared__ float4 s_axc[NA];                            // (ax,ay,az,-25|a|^2)
  __shared__ __align__(16) _Float16 s_b[2][32 * DIM];     // 2 x 16KB tiles
  int tid = threadIdx.x;
  int lane = tid & 31;
  int wave = tid >> 5;
  long long q0 = ((long long)blockIdx.x * 8 + wave) * 16;
  int b = (int)(q0 / NQ);

  const float* ab = anchors + (size_t)b * NA * 3;
  for (int k = tid; k < NA; k += 256) {
    float ax = ab[k * 3], ay = ab[k * 3 + 1], az = ab[k * 3 + 2];
    s_axc[k] = make_float4(ax, ay, az, -VAR_INV * (ax * ax + ay * ay + az * az));
  }

  int m = lane & 15;
  size_t q = (size_t)q0 + m;
  float qx = xyz_q[q * 3 + 0], qy = xyz_q[q * 3 + 1], qz = xyz_q[q * 3 + 2];
  float qc = -VAR_INV * (qx * qx + qy * qy + qz * qz);
  float q2x = 2.0f * VAR_INV * qx, q2y = 2.0f * VAR_INV * qy, q2z = 2.0f * VAR_INV * qz;

  v8f acc[16] = {};
  const uint4* gw = reinterpret_cast<const uint4*>(af_swz + (size_t)b * NA * DIM);
  int kb = (lane & 16) ? 8 : 0;
  float ps0 = 0.0f, ps1 = 0.0f;

#if USE_ASYNC_LDS
  async_stage16k(gw, reinterpret_cast<uint4*>(&s_b[0][0]), tid);
#else
  uint4* sb4 = reinterpret_cast<uint4*>(&s_b[0][0]);
  uint4 pf[4];
#pragma unroll
  for (int i = 0; i < 4; ++i) pf[i] = gw[tid * 4 + i];
#endif

  for (int kt = 0; kt < NA / 32; ++kt) {
#if USE_ASYNC_LDS
    __builtin_amdgcn_s_wait_asynccnt(0);   // tile kt landed in LDS
    __syncthreads();                       // + all waves done reading buf kt+1&1
    if (kt + 1 < NA / 32)
      async_stage16k(gw + (size_t)(kt + 1) * 1024,
                     reinterpret_cast<uint4*>(&s_b[(kt + 1) & 1][0]), tid);
    const _Float16* cur = &s_b[kt & 1][0];
#else
    __syncthreads();                       // previous tile fully consumed
#pragma unroll
    for (int i = 0; i < 4; ++i) sb4[tid * 4 + i] = pf[i];
    if (kt + 1 < NA / 32) {
#pragma unroll
      for (int i = 0; i < 4; ++i)
        pf[i] = gw[(size_t)(kt + 1) * 1024 + tid * 4 + i];
    }
    __syncthreads();                       // stores visible to all waves
    const _Float16* cur = &s_b[0][0];
#endif

    union { v16h h; _Float16 e[16]; } af;
#pragma unroll
    for (int j = 0; j < 8; ++j) {
      int kp = kb + ((j < 4) ? (2 * j) : (8 + 2 * j));
#pragma unroll
      for (int t = 0; t < 2; ++t) {
        float4 a4 = s_axc[kt * 32 + kp + t];
        float arg = qc + a4.w + (q2x * a4.x + q2y * a4.y + q2z * a4.z);
        float w = __expf(arg);
        if (t == 0) ps0 += w; else ps1 += w;
        af.e[2 * j + t] = (_Float16)w;
      }
    }
#pragma unroll
    for (int nt = 0; nt < 16; ++nt) {
      v16h bf = load_bfrag(cur, nt, DIM, lane);
      acc[nt] = __builtin_amdgcn_wmma_f32_16x16x32_f16(
          false, af.h, false, bf, (short)0, acc[nt], false, false);
    }
  }

  // lane L and L^16 hold complementary K halves of the same query
  float psum = ps0 + ps1;
  float total = psum + __shfl_xor(psum, 16, 32);
  float inv[8];
#pragma unroll
  for (int r = 0; r < 8; ++r) {
    float t = __shfl(total, r + ((lane & 16) ? 8 : 0), 32);
    inv[r] = 1.0f / t;
  }

  int n = lane & 15;
  int mrow = (lane & 16) ? 8 : 0;
#pragma unroll
  for (int nt = 0; nt < 16; ++nt) {
#pragma unroll
    for (int r = 0; r < 8; ++r) {
      c_out[((size_t)q0 + mrow + r) * DIM + nt * 16 + n] =
          (_Float16)(acc[nt][r] * inv[r]);
    }
  }
}

// ---------------------------------------------------------------------------
// Phase 2: fully fused decoder.  4 waves/block share staged weight tiles.
//   lat = c@fc0+b ; net = relu(lat)@fc1+b ; 5x resnet blocks ; occ head.
// ---------------------------------------------------------------------------
__global__ __launch_bounds__(128) void decoder_kernel(
    const _Float16* __restrict__ c_h,
    const _Float16* __restrict__ fc0s, const float* __restrict__ fc0_b,
    const _Float16* __restrict__ fc1s, const float* __restrict__ fc1bp,
    const _Float16* __restrict__ fccs, const float* __restrict__ fccbp,
    const _Float16* __restrict__ b0s,  const float* __restrict__ b0bp,
    const _Float16* __restrict__ b1s,  const float* __restrict__ b1bp,
    const float* __restrict__ out_w,   const float* __restrict__ out_b,
    float* __restrict__ occ) {
  __shared__ __align__(16) _Float16 s_w[32 * DIM];       // 16KB staged B tile
  __shared__ __align__(16) _Float16 s_t[4 * 16 * DIM];   // 32KB per-wave slabs
  int tid = threadIdx.x, lane = tid & 31, wave = tid >> 5;
  _Float16* lsw = s_t + (size_t)wave * 16 * DIM;
  long long q0 = ((long long)blockIdx.x * 4 + wave) * 16;
  int lN = lane & 15;
  int mrow = (lane & 16) ? 8 : 0;

  // ---- lat = c @ fc0 + fc0_b  (K=256, N=256) ----
  v16h ca[8];
#pragma unroll
  for (int kt = 0; kt < 8; ++kt)
    ca[kt] = load_afrag(c_h + (size_t)q0 * DIM, DIM, kt * 32, lane);
  v8f lacc[16] = {};
  gemm_staged<8, 16, DIM, 128, false>(fc0s, ca, lacc, s_w, tid, lane);
#pragma unroll
  for (int nt = 0; nt < 16; ++nt) lacc[nt] = lacc[nt] + fc0_b[nt * 16 + lN];

  // stage lat to per-wave LDS slab, reload as resident A-frags
  __syncthreads();
#pragma unroll
  for (int nt = 0; nt < 16; ++nt)
#pragma unroll
    for (int r = 0; r < 8; ++r)
      lsw[(mrow + r) * DIM + nt * 16 + lN] = (_Float16)lacc[nt][r];
  __syncthreads();
  v16h la[8];
#pragma unroll
  for (int kt = 0; kt < 8; ++kt) la[kt] = load_afrag(lsw, DIM, kt * 32, lane);

  // ---- net = relu(lat) @ fc1 + b1  (K=256, N=64p) ----
  v8f nacc[4] = {};
  gemm_staged<8, 4, HP, 128, true>(fc1s, la, nacc, s_w, tid, lane);
#pragma unroll
  for (int nt = 0; nt < 4; ++nt) nacc[nt] = nacc[nt] + fc1bp[nt * 16 + lN];

  // ---- 5 resnet blocks ----
  for (int blk = 0; blk < NBLK; ++blk) {
    const _Float16* fw  = fccs + (size_t)blk * DIM * HP;
    const _Float16* w0  = b0s  + (size_t)blk * HP * HP;
    const _Float16* w1  = b1s  + (size_t)blk * HP * HP;
    const float*    fb  = fccbp + blk * HP;
    const float*    b0b = b0bp  + blk * HP;
    const float*    b1b = b1bp  + blk * HP;

    // net += lat @ fcc + fcc_b
    gemm_staged<8, 4, HP, 128, false>(fw, la, nacc, s_w, tid, lane);
#pragma unroll
    for (int nt = 0; nt < 4; ++nt) nacc[nt] = nacc[nt] + fb[nt * 16 + lN];

    // h = relu(net) @ blk0 + b0  (transpose relu(net) via LDS)
    __syncthreads();
#pragma unroll
    for (int nt = 0; nt < 4; ++nt)
#pragma unroll
      for (int r = 0; r < 8; ++r) {
        float v = nacc[nt][r];
        lsw[(mrow + r) * HP + nt * 16 + lN] = (_Float16)(v > 0.0f ? v : 0.0f);
      }
    __syncthreads();
    v16h na[2];
    na[0] = load_afrag(lsw, HP, 0, lane);
    na[1] = load_afrag(lsw, HP, 32, lane);
    v8f hacc[4] = {};
    gemm_staged<2, 4, HP, 128, false>(w0, na, hacc, s_w, tid, lane);
#pragma unroll
    for (int nt = 0; nt < 4; ++nt) hacc[nt] = hacc[nt] + b0b[nt * 16 + lN];

    // net += relu(h) @ blk1 + b1
    __syncthreads();
#pragma unroll
    for (int nt = 0; nt < 4; ++nt)
#pragma unroll
      for (int r = 0; r < 8; ++r) {
        float v = hacc[nt][r];
        lsw[(mrow + r) * HP + nt * 16 + lN] = (_Float16)(v > 0.0f ? v : 0.0f);
      }
    __syncthreads();
    v16h ha[2];
    ha[0] = load_afrag(lsw, HP, 0, lane);
    ha[1] = load_afrag(lsw, HP, 32, lane);
    gemm_staged<2, 4, HP, 128, false>(w1, ha, nacc, s_w, tid, lane);
#pragma unroll
    for (int nt = 0; nt < 4; ++nt) nacc[nt] = nacc[nt] + b1b[nt * 16 + lN];
  }

  // ---- occ = relu(net) @ out_w + out_b  (first HH columns are real) ----
  __syncthreads();
  float* lsf = reinterpret_cast<float*>(lsw);  // 16*64 f32 fits in 8KB slab
#pragma unroll
  for (int nt = 0; nt < 4; ++nt)
#pragma unroll
    for (int r = 0; r < 8; ++r) {
      float v = nacc[nt][r];
      lsf[(mrow + r) * HP + nt * 16 + lN] = v > 0.0f ? v : 0.0f;
    }
  __syncthreads();
  if (lane < 16) {
    float s = out_b[0];
#pragma unroll
    for (int k = 0; k < HH; ++k) s += lsf[lane * HP + k] * out_w[k];
    occ[(size_t)q0 + lane] = s;
  }
}

// ---------------------------------------------------------------------------
extern "C" void kernel_launch(void* const* d_in, const int* in_sizes, int n_in,
                              void* d_out, int out_size, void* d_ws, size_t ws_size,
                              hipStream_t stream) {
  const float* xyz_q        = (const float*)d_in[0];
  const float* anchors      = (const float*)d_in[1];
  const float* anchor_feats = (const float*)d_in[2];
  const float* fc0_w = (const float*)d_in[3];
  const float* fc0_b = (const float*)d_in[4];
  const float* fc1_w = (const float*)d_in[5];
  const float* fc1_b = (const float*)d_in[6];
  const float* fcc_w = (const float*)d_in[7];
  const float* fcc_b = (const float*)d_in[8];
  const float* blk0_w = (const float*)d_in[9];
  const float* blk0_b = (const float*)d_in[10];
  const float* blk1_w = (const float*)d_in[11];
  const float* blk1_b = (const float*)d_in[12];
  const float* out_w = (const float*)d_in[13];
  const float* out_b = (const float*)d_in[14];
  float* occ = (float*)d_out;

  char* ws = (char*)d_ws;
  size_t off = 0;
  auto take = [&](size_t bytes) -> char* {
    char* p = ws + off;
    off = (off + bytes + 255) & ~(size_t)255;
    return p;
  };
  _Float16* af_swz = (_Float16*)take((size_t)BB * NA * DIM * 2);
  _Float16* fc0s   = (_Float16*)take((size_t)DIM * DIM * 2);
  _Float16* fc1s   = (_Float16*)take((size_t)DIM * HP * 2);
  _Float16* fccs   = (_Float16*)take((size_t)NBLK * DIM * HP * 2);
  _Float16* b0s    = (_Float16*)take((size_t)NBLK * HP * HP * 2);
  _Float16* b1s    = (_Float16*)take((size_t)NBLK * HP * HP * 2);
  float*    fc1bp  = (float*)take(HP * 4);
  float*    fccbp  = (float*)take(NBLK * HP * 4);
  float*    b0bp   = (float*)take(NBLK * HP * 4);
  float*    b1bp   = (float*)take(NBLK * HP * 4);
  _Float16* c_h    = (_Float16*)take((size_t)BB * NQ * DIM * 2);

  // ---- prep: swizzle feats + weights, pad biases ----
  auto swz = [&](const float* src, _Float16* dst, int Ks, int Ns, int Kp, int Np) {
    int n = Kp * Np;
    hipLaunchKernelGGL(swz_pad_kernel, dim3((n + 255) / 256), dim3(256), 0, stream,
                       src, dst, Ks, Ns, Kp, Np);
  };
  for (int b = 0; b < BB; ++b)
    swz(anchor_feats + (size_t)b * NA * DIM, af_swz + (size_t)b * NA * DIM,
        NA, DIM, NA, DIM);
  swz(fc0_w, fc0s, DIM, DIM, DIM, DIM);
  swz(fc1_w, fc1s, DIM, HH, DIM, HP);
  for (int i = 0; i < NBLK; ++i) {
    swz(fcc_w + (size_t)i * DIM * HH, fccs + (size_t)i * DIM * HP, DIM, HH, DIM, HP);
    swz(blk0_w + (size_t)i * HH * HH, b0s + (size_t)i * HP * HP, HH, HH, HP, HP);
    swz(blk1_w + (size_t)i * HH * HH, b1s + (size_t)i * HP * HP, HH, HH, HP, HP);
  }
  hipLaunchKernelGGL(pad_bias_kernel, dim3(1), dim3(HP), 0, stream, fc1_b, fc1bp, HH, HP);
  for (int i = 0; i < NBLK; ++i) {
    hipLaunchKernelGGL(pad_bias_kernel, dim3(1), dim3(HP), 0, stream,
                       fcc_b + i * HH, fccbp + i * HP, HH, HP);
    hipLaunchKernelGGL(pad_bias_kernel, dim3(1), dim3(HP), 0, stream,
                       blk0_b + i * HH, b0bp + i * HP, HH, HP);
    hipLaunchKernelGGL(pad_bias_kernel, dim3(1), dim3(HP), 0, stream,
                       blk1_b + i * HH, b1bp + i * HP, HH, HP);
  }

  // ---- phase 1: interpolation (8 waves/block share staged feats) ----
  hipLaunchKernelGGL(interp_kernel, dim3((BB * NQ) / 128), dim3(256), 0, stream,
                     xyz_q, anchors, af_swz, c_h);

  // ---- phase 2: fused decoder (4 waves/block share staged weights) ----
  hipLaunchKernelGGL(decoder_kernel, dim3((BB * NQ) / 64), dim3(128), 0, stream,
                     c_h, fc0s, fc0_b, fc1s, fc1bp, fccs, fccbp,
                     b0s, b0bp, b1s, b1bp, out_w, out_b, occ);
}
